// DebertaLongLocalAttention_28501402976903
// MI455X (gfx1250) — matrix operand order repaired and checked by reading, not compiled, not run on hardware
//
#include <hip/hip_runtime.h>

// ---------------------------------------------------------------------------
// CDNA5 (gfx1250) wave32 WMMA types
// ---------------------------------------------------------------------------
typedef __attribute__((ext_vector_type(16))) __bf16 v16bf;
typedef __attribute__((ext_vector_type(8)))  float  v8f;

// Sizes for this problem
#define S_LEN   4096
#define HID     1024
#define NHEAD   16
#define HD      64
#define BS      256
#define NB      16
#define PB      256

// ---------------------------------------------------------------------------
// Generic WMMA GEMM:  C[M,N] = A[M,K] * B[K,N] (+ bias[N])
//   A: f32 or bf16 (template), B: f32, C: bf16 or f32 (template)
//   Workgroup tile 128x128, 8 waves (4x2), wave tile 32x64 (2x4 16x16 tiles),
//   K step 32. Double-buffered LDS staging: global loads for tile k+1 are
//   issued before the WMMAs of tile k and stored to the alternate buffer
//   afterwards -> one barrier per K-step, loads hidden under matrix ops.
//   Grid: (N/128, M/128). M, N multiples of 128; K multiple of 32.
// ---------------------------------------------------------------------------
template <typename TA, bool OUT_BF16>
__global__ __launch_bounds__(256)
void wmma_gemm_kernel(const TA* __restrict__ A, int lda,
                      const float* __restrict__ B, int ldb,
                      const float* __restrict__ bias,
                      void* __restrict__ Cout, int ldc, int K)
{
    constexpr int KT = 32;
    __shared__ __bf16 As[2][128][KT + 8];
    __shared__ __bf16 Bs[2][KT][128 + 8];

    const int tid   = threadIdx.x;
    const int wave  = tid >> 5;
    const int lane  = tid & 31;
    const int lrow  = lane & 15;
    const int khalf = lane >> 4;
    const int waveM = wave >> 1;     // 0..3
    const int waveN = wave & 1;      // 0..1
    const int m0 = blockIdx.y * 128;
    const int n0 = blockIdx.x * 128;

    // per-thread staging coordinates (16 contiguous elements each)
    const int arow = tid >> 1;           // 0..127
    const int acol = (tid & 1) * 16;     // 0 or 16
    const int brow = tid >> 3;           // 0..31
    const int bcol = (tid & 7) * 16;     // 0..112

    v8f acc[2][4] = {};
    float ra[16], rb[16];

    // prologue: stage tile 0 into buffer 0
    {
        const TA* sa = A + (size_t)(m0 + arow) * lda + acol;
        const float* sb = B + (size_t)brow * ldb + n0 + bcol;
#pragma unroll
        for (int i = 0; i < 16; ++i) { ra[i] = (float)sa[i]; rb[i] = sb[i]; }
#pragma unroll
        for (int i = 0; i < 16; ++i) {
            As[0][arow][acol + i] = (__bf16)ra[i];
            Bs[0][brow][bcol + i] = (__bf16)rb[i];
        }
    }
    __syncthreads();

    const int nk = K / KT;
    for (int it = 0; it < nk; ++it) {
        const int buf = it & 1;
        const bool has_next = (it + 1) < nk;

        // issue global loads for the next tile (consumed after the WMMAs)
        if (has_next) {
            const int k0 = (it + 1) * KT;
            const TA* sa = A + (size_t)(m0 + arow) * lda + k0 + acol;
            const float* sb = B + (size_t)(k0 + brow) * ldb + n0 + bcol;
#pragma unroll
            for (int i = 0; i < 16; ++i) { ra[i] = (float)sa[i]; rb[i] = sb[i]; }
        }

        // A fragments (16-bit A 16x32 layout: lanes 0-15 K={0..7,16..23},
        // lanes 16-31 K={8..15,24..31}; element j <-> VGPR j/2)
        v16bf af[2];
#pragma unroll
        for (int tm = 0; tm < 2; ++tm) {
            int row = waveM * 32 + tm * 16 + lrow;
#pragma unroll
            for (int j = 0; j < 16; ++j) {
                int kk = j + 8 * ((j >> 3) + khalf);
                af[tm][j] = As[buf][row][kk];
            }
        }
        // B fragments (lane = N, lanes 0-15 K={0..7,16..23}, lanes 16-31 K={8..15,24..31})
        v16bf bfr[4];
#pragma unroll
        for (int tn = 0; tn < 4; ++tn) {
            int col = waveN * 64 + tn * 16 + lrow;
#pragma unroll
            for (int j = 0; j < 16; ++j)
                bfr[tn][j] = Bs[buf][j + 8 * ((j >> 3) + khalf)][col];
        }
#pragma unroll
        for (int tm = 0; tm < 2; ++tm)
#pragma unroll
            for (int tn = 0; tn < 4; ++tn)
                acc[tm][tn] = __builtin_amdgcn_wmma_f32_16x16x32_bf16(
                    false, af[tm], false, bfr[tn], (short)0, acc[tm][tn],
                    false, false);

        // store next tile into the alternate buffer
        if (has_next) {
#pragma unroll
            for (int i = 0; i < 16; ++i) {
                As[buf ^ 1][arow][acol + i] = (__bf16)ra[i];
                Bs[buf ^ 1][brow][bcol + i] = (__bf16)rb[i];
            }
        }
        __syncthreads();
    }

    // epilogue (D layout: VGPR r -> row r + 8*khalf, lane&15 -> col)
#pragma unroll
    for (int tm = 0; tm < 2; ++tm) {
#pragma unroll
        for (int tn = 0; tn < 4; ++tn) {
            int col = n0 + waveN * 64 + tn * 16 + lrow;
            float bv = bias ? bias[col] : 0.0f;
#pragma unroll
            for (int r = 0; r < 8; ++r) {
                int row = m0 + waveM * 32 + tm * 16 + r + 8 * khalf;
                float v = acc[tm][tn][r] + bv;
                if constexpr (OUT_BF16)
                    ((__bf16*)Cout)[(size_t)row * ldc + col] = (__bf16)v;
                else
                    ((float*)Cout)[(size_t)row * ldc + col] = v;
            }
        }
    }
}

// ---------------------------------------------------------------------------
// Bias gather-source vectors (the flattened-take quirk):
//   c2pv[i>=256] = dot(Q[row0,head0], pos_k[i-256, head0])
//   p2cv[i>=256] = dot(pos_q[row0,head0], K[i-256, head0])
//   first 256 entries correspond to the zero pad block -> 0.
// ---------------------------------------------------------------------------
__global__ void bias_vals_kernel(const __bf16* __restrict__ Q,
                                 const __bf16* __restrict__ Kc,
                                 const __bf16* __restrict__ posq,
                                 const __bf16* __restrict__ posk,
                                 float* __restrict__ c2pv,
                                 float* __restrict__ p2cv)
{
    int i = blockIdx.x * blockDim.x + threadIdx.x;
    if (i >= 512) return;
    float c2 = 0.0f, p2 = 0.0f;
    if (i >= 256) {
        int r = i - 256;
#pragma unroll 8
        for (int d = 0; d < HD; ++d) {
            c2 += (float)Q[d] * (float)posk[r * 128 + d];
            p2 += (float)posq[d] * (float)Kc[(size_t)r * HID + d];
        }
    }
    c2pv[i] = c2;
    p2cv[i] = p2;
}

// ---------------------------------------------------------------------------
// Build the shared [256 x 768] bias matrix via the log-bucket position map.
// ---------------------------------------------------------------------------
__global__ void bias_build_kernel(const float* __restrict__ c2pv,
                                  const float* __restrict__ p2cv,
                                  float* __restrict__ biasmat)
{
    int idx = blockIdx.x * blockDim.x + threadIdx.x;
    if (idx >= BS * 3 * BS) return;
    int q = idx / (3 * BS);
    int k = idx - q * (3 * BS);
    int rel = q - k;

    const int max_exact = PB / 2;  // 128
    int sign = (rel > 0) - (rel < 0);
    bool in_exact = (rel < max_exact) && (rel > -max_exact);
    int abs_pos = in_exact ? (max_exact - 1) : (rel < 0 ? -rel : rel);
    int bucket;
    if (abs_pos <= max_exact) {
        bucket = rel;
    } else {
        float x = logf((float)(PB - 1) / (float)max_exact);
        float xx = logf((float)abs_pos / (float)max_exact) / x;
        float log_pos = ceilf(xx * (float)(max_exact - 1)) + (float)max_exact;
        bucket = (int)(log_pos * (float)sign);
    }
    int id = bucket + PB;
    id = min(max(id, 0), 2 * PB - 1);
    biasmat[idx] = c2pv[id] + p2cv[id];
}

// ---------------------------------------------------------------------------
// Local attention, one WG per (block, head). 8 waves, each owns 32 Q rows.
// Flash-style loop over 24 chunks of 32 K-columns. K/V chunks are staged
// into LDS with GLOBAL_LOAD_ASYNC_TO_LDS_B128 (ASYNCcnt path, no VGPR
// round-trip); out-of-range rows (sequence edges) are zero-filled with ds
// stores under EXEC divergence — matches reference: pad columns unmasked,
// bias still applied there.
// ---------------------------------------------------------------------------
__global__ __launch_bounds__(256)
void attn_kernel(const __bf16* __restrict__ Q,
                 const __bf16* __restrict__ Kc,
                 const __bf16* __restrict__ Vc,
                 const float* __restrict__ biasmat,
                 __bf16* __restrict__ ctx)
{
    constexpr int HDp = HD + 8;   // row stride 144B: keeps 16B alignment
    __shared__ __bf16 Ks[32][HDp];
    __shared__ __bf16 Vs[32][HDp];
    __shared__ __bf16 Pw[8][32][32 + 8];

    const int blk  = blockIdx.x;
    const int head = blockIdx.y;
    const int tid  = threadIdx.x;
    const int wave = tid >> 5;
    const int lane = tid & 31;
    const int lrow = lane & 15;
    const int khalf = lane >> 4;
    const int hc = head * HD;
    const int qrow0 = blk * BS + wave * 32;
    const int k3row0 = blk * BS - BS;

    // staging coordinates: 8 bf16 (16 bytes) per thread
    const int srow = tid >> 3;          // 0..31
    const int scol = (tid & 7) * 8;     // 0..56

    // Q A-fragments: 2 m-tiles x 2 k-steps (head dim 64)
    v16bf qf[2][2];
#pragma unroll
    for (int tm = 0; tm < 2; ++tm) {
        const __bf16* qp = Q + (size_t)(qrow0 + tm * 16 + lrow) * HID + hc;
#pragma unroll
        for (int ks = 0; ks < 2; ++ks)
#pragma unroll
            for (int j = 0; j < 16; ++j) {
                int kk = ks * 32 + j + 8 * ((j >> 3) + khalf);
                qf[tm][ks][j] = qp[kk];
            }
    }

    float mrow[2][8], lsum[2][8];
#pragma unroll
    for (int tm = 0; tm < 2; ++tm)
#pragma unroll
        for (int r = 0; r < 8; ++r) { mrow[tm][r] = -1e30f; lsum[tm][r] = 0.0f; }
    v8f oacc[2][4] = {};

    const float inv_scale = 0.07216878364870323f;  // 1/sqrt(HD*3)

    for (int c = 0; c < 24; ++c) {
        __syncthreads();
        // stage K/V chunk: 32 rows x 64 dims, one async b128 per thread
        {
            int g = k3row0 + c * 32 + srow;
            if (g >= 0 && g < S_LEN) {
                unsigned ldsK = (unsigned)(size_t)&Ks[srow][scol];
                unsigned ldsV = (unsigned)(size_t)&Vs[srow][scol];
                unsigned long long ga =
                    (unsigned long long)(Kc + (size_t)g * HID + hc + scol);
                unsigned long long gb =
                    (unsigned long long)(Vc + (size_t)g * HID + hc + scol);
                asm volatile("global_load_async_to_lds_b128 %0, %1, off"
                             :: "v"(ldsK), "v"(ga) : "memory");
                asm volatile("global_load_async_to_lds_b128 %0, %1, off"
                             :: "v"(ldsV), "v"(gb) : "memory");
            } else {
#pragma unroll
                for (int i = 0; i < 8; ++i) {
                    Ks[srow][scol + i] = (__bf16)0.0f;
                    Vs[srow][scol + i] = (__bf16)0.0f;
                }
            }
            asm volatile("s_wait_asynccnt 0x0" ::: "memory");
        }
        __syncthreads();

        // bias tile loads first so they overlap the score WMMAs
        float bb[2][2][8];
#pragma unroll
        for (int tm = 0; tm < 2; ++tm)
#pragma unroll
            for (int tn = 0; tn < 2; ++tn) {
                int colg = c * 32 + tn * 16 + lrow;
#pragma unroll
                for (int r = 0; r < 8; ++r) {
                    int rq = wave * 32 + tm * 16 + r + 8 * khalf;
                    bb[tm][tn][r] = biasmat[rq * (3 * BS) + colg];
                }
            }

        // S = Q * K3^T for 32x32 chunk
        float sv[2][2][8];
#pragma unroll
        for (int tm = 0; tm < 2; ++tm)
#pragma unroll
            for (int tn = 0; tn < 2; ++tn) {
                v8f s = {};
#pragma unroll
                for (int ks = 0; ks < 2; ++ks) {
                    v16bf kf;
                    int n = tn * 16 + lrow;  // K3 row within chunk
#pragma unroll
                    for (int j = 0; j < 16; ++j)
                        kf[j] = Ks[n][ks * 32 + j + 8 * ((j >> 3) + khalf)];
                    s = __builtin_amdgcn_wmma_f32_16x16x32_bf16(
                        false, qf[tm][ks], false, kf, (short)0, s, false, false);
                }
#pragma unroll
                for (int r = 0; r < 8; ++r)
                    sv[tm][tn][r] = (s[r] + bb[tm][tn][r]) * inv_scale;
            }

        // online softmax (row spread over 16 lanes of each half-wave)
#pragma unroll
        for (int tm = 0; tm < 2; ++tm) {
#pragma unroll
            for (int r = 0; r < 8; ++r) {
                float m = fmaxf(sv[tm][0][r], sv[tm][1][r]);
                for (int off = 8; off >= 1; off >>= 1)
                    m = fmaxf(m, __shfl_xor(m, off, 32));
                float mold = mrow[tm][r];
                float mnew = fmaxf(mold, m);
                float corr = __expf(mold - mnew);
                float p0 = __expf(sv[tm][0][r] - mnew);
                float p1 = __expf(sv[tm][1][r] - mnew);
                float ps = p0 + p1;
                for (int off = 8; off >= 1; off >>= 1)
                    ps += __shfl_xor(ps, off, 32);
                lsum[tm][r] = lsum[tm][r] * corr + ps;
                mrow[tm][r] = mnew;
                sv[tm][0][r] = p0;
                sv[tm][1][r] = p1;
#pragma unroll
                for (int vn = 0; vn < 4; ++vn)
                    oacc[tm][vn][r] *= corr;
            }
            // P -> per-wave LDS scratch (D layout write)
#pragma unroll
            for (int tn = 0; tn < 2; ++tn)
#pragma unroll
                for (int r = 0; r < 8; ++r)
                    Pw[wave][tm * 16 + r + 8 * khalf][tn * 16 + lrow] =
                        (__bf16)sv[tm][tn][r];
        }

        // ctx += P (A-frag from LDS) x V (B-frag from LDS)
        v16bf vf[4];
#pragma unroll
        for (int vn = 0; vn < 4; ++vn)
#pragma unroll
            for (int j = 0; j < 16; ++j)
                vf[vn][j] = Vs[j + 8 * ((j >> 3) + khalf)][vn * 16 + lrow];
#pragma unroll
        for (int tm = 0; tm < 2; ++tm) {
            v16bf pf;
#pragma unroll
            for (int j = 0; j < 16; ++j) {
                int kk = j + 8 * ((j >> 3) + khalf);
                pf[j] = Pw[wave][tm * 16 + lrow][kk];
            }
#pragma unroll
            for (int vn = 0; vn < 4; ++vn)
                oacc[tm][vn] = __builtin_amdgcn_wmma_f32_16x16x32_bf16(
                    false, pf, false, vf[vn], (short)0, oacc[tm][vn],
                    false, false);
        }
    }

    // normalize and store ctx (bf16)
#pragma unroll
    for (int tm = 0; tm < 2; ++tm)
#pragma unroll
        for (int vn = 0; vn < 4; ++vn)
#pragma unroll
            for (int r = 0; r < 8; ++r) {
                int row = qrow0 + tm * 16 + r + 8 * khalf;
                int col = hc + vn * 16 + lrow;
                float v = oacc[tm][vn][r] / lsum[tm][r];
                ctx[(size_t)row * HID + col] = (__bf16)v;
            }
}

// ---------------------------------------------------------------------------
// Residual + LayerNorm: out = LN(oproj + hidden) * scale + bias
// ---------------------------------------------------------------------------
__global__ __launch_bounds__(256)
void ln_kernel(const float* __restrict__ oproj,
               const float* __restrict__ hidden,
               const float* __restrict__ lnsc,
               const float* __restrict__ lnbi,
               float* __restrict__ out)
{
    __shared__ float red[256];
    const int row = blockIdx.x;
    const int tid = threadIdx.x;
    const float* op = oproj + (size_t)row * HID;
    const float* hp = hidden + (size_t)row * HID;

    float x[4];
    float s = 0.0f;
#pragma unroll
    for (int i = 0; i < 4; ++i) {
        x[i] = op[tid + 256 * i] + hp[tid + 256 * i];
        s += x[i];
    }
    red[tid] = s;
    __syncthreads();
    for (int st = 128; st >= 1; st >>= 1) {
        if (tid < st) red[tid] += red[tid + st];
        __syncthreads();
    }
    float mu = red[0] / (float)HID;
    __syncthreads();

    float s2 = 0.0f;
#pragma unroll
    for (int i = 0; i < 4; ++i) {
        float d = x[i] - mu;
        s2 += d * d;
    }
    red[tid] = s2;
    __syncthreads();
    for (int st = 128; st >= 1; st >>= 1) {
        if (tid < st) red[tid] += red[tid + st];
        __syncthreads();
    }
    float var = red[0] / (float)HID;
    float rs = rsqrtf(var + 1e-7f);
#pragma unroll
    for (int i = 0; i < 4; ++i) {
        int cidx = tid + 256 * i;
        out[(size_t)row * HID + cidx] = (x[i] - mu) * rs * lnsc[cidx] + lnbi[cidx];
    }
}

// ---------------------------------------------------------------------------
extern "C" void kernel_launch(void* const* d_in, const int* in_sizes, int n_in,
                              void* d_out, int out_size, void* d_ws, size_t ws_size,
                              hipStream_t stream)
{
    (void)in_sizes; (void)n_in; (void)out_size; (void)ws_size;

    const float* hidden = (const float*)d_in[0];
    const float* relpos = (const float*)d_in[1];
    const float* Wq = (const float*)d_in[2];
    const float* bq = (const float*)d_in[3];
    const float* Wk = (const float*)d_in[4];
    const float* bk = (const float*)d_in[5];
    const float* Wv = (const float*)d_in[6];
    const float* bv = (const float*)d_in[7];
    const float* Wo = (const float*)d_in[8];
    const float* bo = (const float*)d_in[9];
    const float* lnsc = (const float*)d_in[10];
    const float* lnbi = (const float*)d_in[11];
    float* out = (float*)d_out;

    // workspace carve-up (~49 MB)
    char* ws = (char*)d_ws;
    const size_t SH2 = (size_t)S_LEN * HID * sizeof(__bf16);   // 8 MB
    __bf16* qb    = (__bf16*)(ws);               ws += SH2;
    __bf16* kb    = (__bf16*)(ws);               ws += SH2;
    __bf16* vb    = (__bf16*)(ws);               ws += SH2;
    __bf16* ctxb  = (__bf16*)(ws);               ws += SH2;
    float*  oproj = (float*)(ws);                ws += (size_t)S_LEN * HID * sizeof(float);
    __bf16* poskb = (__bf16*)(ws);               ws += (size_t)256 * 128 * sizeof(__bf16);
    __bf16* posqb = (__bf16*)(ws);               ws += (size_t)128 * 128 * sizeof(__bf16);
    float*  c2pv  = (float*)(ws);                ws += 512 * sizeof(float);
    float*  p2cv  = (float*)(ws);                ws += 512 * sizeof(float);
    float*  biasmat = (float*)(ws);              ws += (size_t)BS * 3 * BS * sizeof(float);

    // 1) QKV projections (f32 in, bf16 out)
    wmma_gemm_kernel<float, true><<<dim3(HID / 128, S_LEN / 128), 256, 0, stream>>>(
        hidden, HID, Wq, HID, bq, qb, HID, HID);
    wmma_gemm_kernel<float, true><<<dim3(HID / 128, S_LEN / 128), 256, 0, stream>>>(
        hidden, HID, Wk, HID, bk, kb, HID, HID);
    wmma_gemm_kernel<float, true><<<dim3(HID / 128, S_LEN / 128), 256, 0, stream>>>(
        hidden, HID, Wv, HID, bv, vb, HID, HID);

    // 2) position projection slices (only rows/head-0 cols feed the bias)
    wmma_gemm_kernel<float, true><<<dim3(1, 2), 256, 0, stream>>>(
        relpos, HID, Wk, HID, bk, poskb, 128, HID);   // pos_k rows 0..255
    wmma_gemm_kernel<float, true><<<dim3(1, 1), 256, 0, stream>>>(
        relpos, HID, Wq, HID, bq, posqb, 128, HID);   // pos_q row 0

    // 3) bias vectors + [256,768] bias matrix (flattened-take semantics)
    bias_vals_kernel<<<2, 256, 0, stream>>>(qb, kb, posqb, poskb, c2pv, p2cv);
    bias_build_kernel<<<(BS * 3 * BS) / 256, 256, 0, stream>>>(c2pv, p2cv, biasmat);

    // 4) blocked local attention with WMMA + online softmax + async LDS staging
    attn_kernel<<<dim3(NB, NHEAD), 256, 0, stream>>>(qb, kb, vb, biasmat, ctxb);

    // 5) output projection (bf16 in, f32 out)
    wmma_gemm_kernel<__bf16, false><<<dim3(HID / 128, S_LEN / 128), 256, 0, stream>>>(
        ctxb, HID, Wo, HID, bo, oproj, HID, HID);

    // 6) residual + layernorm
    ln_kernel<<<S_LEN, 256, 0, stream>>>(oproj, hidden, lnsc, lnbi, out);
}
